// MoeLayer_60112362275165
// MI455X (gfx1250) — matrix operand order repaired and checked by reading, not compiled, not run on hardware
//
#include <hip/hip_runtime.h>
#include <hip/hip_bf16.h>

#define TOKENS     4096
#define EMBED      1024
#define HIDDEN     4096
#define NEXPERT    8
#define TILE_M     16
#define HCHUNK     128

typedef __attribute__((ext_vector_type(16))) __bf16 v16bf;
typedef __attribute__((ext_vector_type(8)))  float  v8f;

#if defined(__has_builtin)
#  if __has_builtin(__builtin_amdgcn_global_load_async_to_lds_b128) && \
      __has_builtin(__builtin_amdgcn_s_wait_asynccnt)
#    define USE_ASYNC_COPY 1
#  endif
#endif
#ifndef USE_ASYNC_COPY
#  define USE_ASYNC_COPY 0
#endif

#if USE_ASYNC_COPY
// Builtin signature (from hipcc diagnostic): param0 is
// '__attribute__((__vector_size__(4*sizeof(int)))) int __device__ *'  (AS1),
// param1 is the LDS-side pointer (AS3), then imm offset, imm cpol.
typedef __attribute__((vector_size(4 * sizeof(int)))) int v4i_vec;
typedef __attribute__((address_space(1))) v4i_vec* gv4i_p;
typedef __attribute__((address_space(3))) v4i_vec* lv4i_p;
#endif

// ---- helpers -------------------------------------------------------------

__device__ __forceinline__ unsigned short f2bf(float f) {
    unsigned u = __float_as_uint(f);
    unsigned r = u + 0x7FFFu + ((u >> 16) & 1u);   // round-to-nearest-even
    return (unsigned short)(r >> 16);
}

union V16U { unsigned short u[16]; v16bf v; uint4 q[2]; };

// A-matrix (16x32 bf16) from an LDS row-major [16][row_stride] buffer.
// ISA layout: lane L: M = L&15; elems 0..7 = K kb+off..+7, elems 8..15 =
// K kb+16+off.., off = (L<16)?0:8.
__device__ __forceinline__ v16bf load_a_lds(const unsigned short* mat,
                                            int row_stride, int kb, int lane) {
    V16U a;
    const int m   = lane & 15;
    const int off = (lane < 16) ? 0 : 8;
    const unsigned short* p = mat + m * row_stride + kb + off;
    a.q[0] = *(const uint4*)(p);        // 8 bf16
    a.q[1] = *(const uint4*)(p + 16);   // 8 bf16
    return a.v;
}

// B-matrix (32x16 bf16) from a pre-transposed bf16 weight matrix Bt[N][K].
// ISA layout: lane L: N = nb + (L&15); elems 0..7 = K kb+off..+7, elems 8..15 = K kb+16+off.., off = (L<16)?0:8.
// K is contiguous along the row -> two 16B vector loads per lane.
__device__ __forceinline__ v16bf load_b_bf16t(const unsigned short* __restrict__ Bt,
                                              long ldk, int kb, int nb, int lane) {
    V16U b;
    const unsigned short* p =
        Bt + (long)(nb + (lane & 15)) * ldk + kb + ((lane < 16) ? 0 : 8);
    __builtin_prefetch(p + 256, 0, 1);      // ~8 K-steps ahead (global_prefetch_b8)
    b.q[0] = *(const uint4*)(p);
    b.q[1] = *(const uint4*)(p + 16);
    return b.v;
}

__device__ __forceinline__ v8f wmma_bf16(v16bf a, v16bf b, v8f c) {
    return __builtin_amdgcn_wmma_f32_16x16x32_bf16(
        false, a, false, b, (short)0, c, false, false);
}

// ---- pre-pass kernels ----------------------------------------------------

__global__ void init_counts_kernel(int* counts) {
    if (threadIdx.x < NEXPERT) counts[threadIdx.x] = 0;
}

// x fp32 -> bf16, flat copy
__global__ __launch_bounds__(256) void cvt_x_kernel(
    const float* __restrict__ x, unsigned short* __restrict__ xb, int n)
{
    const int i = blockIdx.x * 256 + threadIdx.x;
    if (i < n) xb[i] = f2bf(x[i]);
}

// Per-expert [K][N] fp32 -> [N][K] bf16 (32x32 LDS tile transpose).
__global__ __launch_bounds__(256) void transpose_bf16_kernel(
    const float* __restrict__ src, unsigned short* __restrict__ dst,
    int K, int N)
{
    __shared__ unsigned short tile[32][33];
    const long eoff = (long)blockIdx.z * K * N;
    const int k0 = blockIdx.y * 32, n0 = blockIdx.x * 32;
    const int tx = threadIdx.x & 31, ty = threadIdx.x >> 5;   // ty: 0..7
#pragma unroll
    for (int i = 0; i < 4; ++i) {
        const int k = ty + i * 8;
        tile[k][tx] = f2bf(src[eoff + (long)(k0 + k) * N + n0 + tx]);
    }
    __syncthreads();
#pragma unroll
    for (int i = 0; i < 4; ++i) {
        const int n = ty + i * 8;
        dst[eoff + (long)(n0 + n) * K + k0 + tx] = tile[tx][n];
    }
}

// ---- gating --------------------------------------------------------------

__global__ __launch_bounds__(128) void gate_kernel(
    const float* __restrict__ x, const float* __restrict__ Wg,
    int* counts, int* tids, int* tks, float* tw)
{
    const int t = blockIdx.x * 128 + threadIdx.x;
    if (t >= TOKENS) return;

    float logit[NEXPERT];
#pragma unroll
    for (int e = 0; e < NEXPERT; ++e) logit[e] = 0.f;

    const float* xr = x + (long)t * EMBED;
    for (int d = 0; d < EMBED; ++d) {
        const float xv = xr[d];
#pragma unroll
        for (int e = 0; e < NEXPERT; ++e) logit[e] += xv * Wg[d * NEXPERT + e];
    }

    // top-2 (ties -> lowest index, matching jax.lax.top_k)
    int e0 = 0; float v0 = logit[0];
#pragma unroll
    for (int e = 1; e < NEXPERT; ++e) if (logit[e] > v0) { v0 = logit[e]; e0 = e; }
    int e1 = -1; float v1 = -3.402823466e38f;
#pragma unroll
    for (int e = 0; e < NEXPERT; ++e)
        if (e != e0 && logit[e] > v1) { v1 = logit[e]; e1 = e; }

    const float w0 = 1.f / (1.f + __expf(v1 - v0));   // softmax over top-2
    const float w1 = 1.f - w0;

    int s0 = atomicAdd(&counts[e0], 1);
    tids[e0 * TOKENS + s0] = t; tks[e0 * TOKENS + s0] = 0; tw[e0 * TOKENS + s0] = w0;
    int s1 = atomicAdd(&counts[e1], 1);
    tids[e1 * TOKENS + s1] = t; tks[e1 * TOKENS + s1] = 1; tw[e1 * TOKENS + s1] = w1;
}

// ---- expert FFN (WMMA) ---------------------------------------------------

__global__ __launch_bounds__(256) void ffn_kernel(
    const unsigned short* __restrict__ xb,    // [TOKENS][1024] bf16
    const unsigned short* __restrict__ W1t,   // [E][4096][1024] bf16 (N,K)
    const unsigned short* __restrict__ W3t,   // [E][4096][1024] bf16 (N,K)
    const unsigned short* __restrict__ W2t,   // [E][1024][4096] bf16 (N,K)
    const int* __restrict__ counts, const int* __restrict__ tids,
    const int* __restrict__ tks,    const float* __restrict__ tw,
    float* __restrict__ contrib)              // [2][TOKENS][1024]
{
    const int e   = blockIdx.y;
    const int cnt = counts[e];
    const int t0  = blockIdx.x * TILE_M;
    if (t0 >= cnt) return;          // block-uniform exit: EXEC stays all-ones

    __shared__ __align__(16) unsigned short xs[TILE_M * EMBED];   // 32 KB
    __shared__ __align__(16) unsigned short hs[TILE_M * HCHUNK];  //  4 KB
    __shared__ int   s_tid[TILE_M];
    __shared__ int   s_tk [TILE_M];
    __shared__ float s_tw [TILE_M];

    const int tidx = threadIdx.x;
    const int lane = tidx & 31;
    const int wave = tidx >> 5;

    if (tidx < TILE_M) {
        const int slot = t0 + tidx;
        if (slot < cnt) {
            s_tid[tidx] = tids[e * TOKENS + slot];
            s_tk [tidx] = tks [e * TOKENS + slot];
            s_tw [tidx] = tw  [e * TOKENS + slot];
        } else { s_tid[tidx] = -1; s_tk[tidx] = 0; s_tw[tidx] = 0.f; }
    }
    __syncthreads();

    // Stage 16 x 1024 bf16 x-tile into LDS (async global->LDS when available).
    for (int i = tidx; i < TILE_M * EMBED / 8; i += 256) {     // 2048 x 16B
        const int m = i >> 7, c = i & 127;
        const int tok = s_tid[m];
#if USE_ASYNC_COPY
        if (tok >= 0) {
            const unsigned short* gsrc = xb + (long)tok * EMBED + (c << 3);
            // generic LDS addr low 32 bits == LDS offset (flat aperture rule)
            __builtin_amdgcn_global_load_async_to_lds_b128(
                (gv4i_p)(unsigned long long)(const void*)gsrc,
                (lv4i_p)(unsigned long long)(unsigned)(unsigned long long)
                    (void*)(xs + (i << 3)),
                0, 0);
        } else {
            ((uint4*)xs)[i] = make_uint4(0u, 0u, 0u, 0u);
        }
#else
        uint4 v = make_uint4(0u, 0u, 0u, 0u);
        if (tok >= 0) v = ((const uint4*)(xb + (long)tok * EMBED))[c];
        ((uint4*)xs)[i] = v;
#endif
    }
#if USE_ASYNC_COPY
    __builtin_amdgcn_s_wait_asynccnt(0);
#endif
    __syncthreads();

    const unsigned short* W1e = W1t + (long)e * HIDDEN * EMBED;
    const unsigned short* W3e = W3t + (long)e * HIDDEN * EMBED;
    const unsigned short* W2e = W2t + (long)e * EMBED * HIDDEN;

    v8f outAcc[8];
#pragma unroll
    for (int i = 0; i < 8; ++i) outAcc[i] = (v8f){0,0,0,0,0,0,0,0};

#pragma unroll 1
    for (int hc = 0; hc < HIDDEN; hc += HCHUNK) {
        // ---- Phase A: wave computes h tile [16 x 16] at cols hc+wave*16
        const int col = hc + wave * 16;
        v8f acc1 = (v8f){0,0,0,0,0,0,0,0};
        v8f acc3 = (v8f){0,0,0,0,0,0,0,0};
#pragma unroll 2
        for (int kb = 0; kb < EMBED; kb += 32) {
            const v16bf a  = load_a_lds(xs, EMBED, kb, lane);
            const v16bf b1 = load_b_bf16t(W1e, EMBED, kb, col, lane);
            acc1 = wmma_bf16(a, b1, acc1);
            const v16bf b3 = load_b_bf16t(W3e, EMBED, kb, col, lane);
            acc3 = wmma_bf16(a, b3, acc3);
        }

        __syncthreads();   // previous chunk's hs fully consumed
        {
            const int m0 = (lane < 16) ? 0 : 8;
            const int n  = lane & 15;
#pragma unroll
            for (int j = 0; j < 8; ++j) {
                const float g = acc1[j];
                // silu via v_rcp_f32 instead of IEEE divide expansion
                const float s = g * __builtin_amdgcn_rcpf(1.f + __expf(-g));
                hs[(m0 + j) * HCHUNK + wave * 16 + n] = f2bf(s * acc3[j]);
            }
        }
        __syncthreads();   // hs chunk visible to all waves

        // ---- Phase B: out[16 x 128 cols of this wave] += h_chunk @ W2_chunk
#pragma unroll 1
        for (int kb = 0; kb < HCHUNK; kb += 32) {
            const v16bf ah = load_a_lds(hs, HCHUNK, kb, lane);
#pragma unroll
            for (int tcol = 0; tcol < 8; ++tcol) {
                const v16bf b2 = load_b_bf16t(W2e, HIDDEN, hc + kb,
                                              wave * 128 + tcol * 16, lane);
                outAcc[tcol] = wmma_bf16(ah, b2, outAcc[tcol]);
            }
        }
    }

    // Scale by routing weight; write per-(k,token) contribution (no atomics).
    {
        const int m0 = (lane < 16) ? 0 : 8;
        const int n  = lane & 15;
#pragma unroll
        for (int tcol = 0; tcol < 8; ++tcol) {
            const int colg = wave * 128 + tcol * 16 + n;
#pragma unroll
            for (int j = 0; j < 8; ++j) {
                const int m   = m0 + j;
                const int tok = s_tid[m];
                if (tok >= 0) {
                    contrib[((long)s_tk[m] * TOKENS + tok) * EMBED + colg] =
                        s_tw[m] * outAcc[tcol][j];
                }
            }
        }
    }
}

__global__ __launch_bounds__(256) void combine_kernel(
    const float* __restrict__ contrib, float* __restrict__ out, int n)
{
    const int i = blockIdx.x * 256 + threadIdx.x;
    if (i < n) out[i] = contrib[i] + contrib[(long)n + i];
}

// ---- launch --------------------------------------------------------------

extern "C" void kernel_launch(void* const* d_in, const int* in_sizes, int n_in,
                              void* d_out, int out_size, void* d_ws, size_t ws_size,
                              hipStream_t stream) {
    const float* x  = (const float*)d_in[0];   // [2,2048,1024]
    const float* Wg = (const float*)d_in[1];   // [1024,8]
    const float* W1 = (const float*)d_in[2];   // [8,1024,4096]
    const float* W3 = (const float*)d_in[3];   // [8,1024,4096]
    const float* W2 = (const float*)d_in[4];   // [8,4096,1024]
    float* out = (float*)d_out;

    char* ws = (char*)d_ws;
    size_t off = 0;
    int* counts = (int*)(ws + off);             off += 256;
    int* tids   = (int*)(ws + off);             off += (size_t)NEXPERT * TOKENS * 4;
    int* tks    = (int*)(ws + off);             off += (size_t)NEXPERT * TOKENS * 4;
    float* tw   = (float*)(ws + off);           off += (size_t)NEXPERT * TOKENS * 4;
    float* contrib = (float*)(ws + off);        off += (size_t)2 * TOKENS * EMBED * 4;
    unsigned short* xb  = (unsigned short*)(ws + off); off += (size_t)TOKENS * EMBED * 2;
    unsigned short* W1t = (unsigned short*)(ws + off); off += (size_t)NEXPERT * EMBED * HIDDEN * 2;
    unsigned short* W3t = (unsigned short*)(ws + off); off += (size_t)NEXPERT * EMBED * HIDDEN * 2;
    unsigned short* W2t = (unsigned short*)(ws + off); off += (size_t)NEXPERT * HIDDEN * EMBED * 2;

    init_counts_kernel<<<1, 32, 0, stream>>>(counts);

    // One-time fp32 -> bf16 conversion / transpose pre-pass.
    cvt_x_kernel<<<(TOKENS * EMBED + 255) / 256, 256, 0, stream>>>(
        x, xb, TOKENS * EMBED);
    {
        dim3 g1(HIDDEN / 32, EMBED / 32, NEXPERT);   // W1,W3: [1024][4096] -> [4096][1024]
        transpose_bf16_kernel<<<g1, 256, 0, stream>>>(W1, W1t, EMBED, HIDDEN);
        transpose_bf16_kernel<<<g1, 256, 0, stream>>>(W3, W3t, EMBED, HIDDEN);
        dim3 g2(EMBED / 32, HIDDEN / 32, NEXPERT);   // W2: [4096][1024] -> [1024][4096]
        transpose_bf16_kernel<<<g2, 256, 0, stream>>>(W2, W2t, HIDDEN, EMBED);
    }

    gate_kernel<<<TOKENS / 128, 128, 0, stream>>>(x, Wg, counts, tids, tks, tw);

    dim3 grid(TOKENS / TILE_M, NEXPERT);
    ffn_kernel<<<grid, 256, 0, stream>>>(xb, W1t, W3t, W2t,
                                         counts, tids, tks, tw, contrib);

    combine_kernel<<<(out_size + 255) / 256, 256, 0, stream>>>(contrib, out, out_size);
}
